// Non_local_79345225826758
// MI455X (gfx1250) — matrix operand order repaired and checked by reading, not compiled, not run on hardware
//
#include <hip/hip_runtime.h>
#include <hip/hip_bf16.h>

typedef __attribute__((ext_vector_type(16))) _Float16 v16h;
typedef __attribute__((ext_vector_type(8)))  _Float16 v8h;
typedef __attribute__((ext_vector_type(8)))  float    v8f;

#define HW 4096
#define CC 256
#define NB 8
#define MTOT (NB * HW)   // 32768

__device__ __forceinline__ v8f wmma16(v16h a, v16h b, v8f c) {
    // D = A(16x32 f16) * B(32x16 f16) + C(16x16 f32)
    return __builtin_amdgcn_wmma_f32_16x16x32_f16(
        false, a, false, b, (short)0, c, false, false);
}

// ---- fragment loaders (wave32, same contiguous k order k=16h+s in A and B) ----------------
// A 16x32 f16: lane<16 holds row r=lane, K in k0+0..15;
//              lane>=16 same row, K in k0+16..31.
__device__ __forceinline__ v16h load_a_f16(const _Float16* A, int lda,
                                           int m0, int k0, int lane) {
    const int h = (lane >> 4) & 1, r = lane & 15;
    const _Float16* row = A + (size_t)(m0 + r) * lda + k0;
    v8h lo = *(const v8h*)(row + h * 16);
    v8h hi = *(const v8h*)(row + h * 16 + 8);
    return __builtin_shufflevector(lo, hi, 0,1,2,3,4,5,6,7,8,9,10,11,12,13,14,15);
}

__device__ __forceinline__ v16h load_a_f32cvt(const float* A, int lda,
                                              int m0, int k0, int lane) {
    const int h = (lane >> 4) & 1, r = lane & 15;
    const float* row = A + (size_t)(m0 + r) * lda + k0;
    const float4* p0 = (const float4*)(row + h * 16);
    const float4* p1 = (const float4*)(row + h * 16 + 8);
    float4 a0 = p0[0], a1 = p0[1], b0 = p1[0], b1 = p1[1];
    float f[16] = {a0.x,a0.y,a0.z,a0.w, a1.x,a1.y,a1.z,a1.w,
                   b0.x,b0.y,b0.z,b0.w, b1.x,b1.y,b1.z,b1.w};
    v16h v;
#pragma unroll
    for (int j = 0; j < 16; ++j) v[j] = (_Float16)f[j];
    return v;
}

// B 32x16 f16 where B[k][n] = T[n][k], T row-major [N x K]:
// lane c = lane&15 -> column n0+c; lanes<16 K=k0..k0+15, lanes>=16 K=k0+16..+31.
__device__ __forceinline__ v16h load_bT_f16(const _Float16* T, int ldt,
                                            int n0, int k0, int lane) {
    const int h = (lane >> 4) & 1, c = lane & 15;
    return *(const v16h*)(T + (size_t)(n0 + c) * ldt + k0 + h * 16);
}

__device__ __forceinline__ v16h load_bT_f32cvt(const float* T, int ldt,
                                               int n0, int k0, int lane) {
    const int h = (lane >> 4) & 1, c = lane & 15;
    const float4* p = (const float4*)(T + (size_t)(n0 + c) * ldt + k0 + h * 16);
    float4 a0 = p[0], a1 = p[1], a2 = p[2], a3 = p[3];
    float f[16] = {a0.x,a0.y,a0.z,a0.w, a1.x,a1.y,a1.z,a1.w,
                   a2.x,a2.y,a2.z,a2.w, a3.x,a3.y,a3.z,a3.w};
    v16h v;
#pragma unroll
    for (int j = 0; j < 16; ++j) v[j] = (_Float16)f[j];
    return v;
}

// ---- projection kernel: Y = X(M x 256) * W^T(256 x 256) + b ---------------
// 512 threads = 16 waves; wave w owns N-tile w*16; block owns 16 rows.
// MODE 0: L2-normalize rows, store f16 row-major (qn16/kn16)
// MODE 1: store f32 (vp out) + f16 transposed [B][C][HW] (vpT)
// MODE 2: store f32 = Y + residual (final output)
template <int MODE>
__global__ void __launch_bounds__(512)
proj_kernel(const float* __restrict__ X,
            const float* __restrict__ W,
            const float* __restrict__ bias,
            _Float16* __restrict__ out16,
            float* __restrict__ out32,
            const float* __restrict__ resid) {
    const int tid  = threadIdx.x;
    const int wave = tid >> 5, lane = tid & 31;
    const int h = (lane >> 4) & 1, c = lane & 15;
    const int m0 = blockIdx.x * 16;
    const int n0 = wave * 16;

    v8f acc = {};
#pragma unroll
    for (int k0 = 0; k0 < CC; k0 += 32) {
        v16h a = load_a_f32cvt(X, CC, m0, k0, lane);
        v16h b = load_bT_f32cvt(W, CC, n0, k0, lane);
        acc = wmma16(a, b, acc);
    }
    const float bn = bias[n0 + c];
    float y[8];
#pragma unroll
    for (int v = 0; v < 8; ++v) y[v] = acc[v] + bn;

    if (MODE == 0) {
        __shared__ float rowss[16];
        if (tid < 16) rowss[tid] = 0.f;
        __syncthreads();
        float s[8];
#pragma unroll
        for (int v = 0; v < 8; ++v) s[v] = y[v] * y[v];
#pragma unroll
        for (int m = 1; m <= 8; m <<= 1)
#pragma unroll
            for (int v = 0; v < 8; ++v) s[v] += __shfl_xor(s[v], m, 32);
        if (c == 0) {
#pragma unroll
            for (int v = 0; v < 8; ++v) atomicAdd(&rowss[v + 8 * h], s[v]);
        }
        __syncthreads();
#pragma unroll
        for (int v = 0; v < 8; ++v) {
            const float inv = rsqrtf(rowss[v + 8 * h] + 1e-20f);
            const int m = m0 + v + 8 * h;
            out16[(size_t)m * CC + n0 + c] = (_Float16)(y[v] * inv);
        }
    } else if (MODE == 1) {
#pragma unroll
        for (int v = 0; v < 8; ++v) {
            const int m = m0 + v + 8 * h;
            out32[(size_t)m * CC + n0 + c] = y[v];
            const int b = m >> 12, t = m & (HW - 1);
            out16[((size_t)b * CC + n0 + c) * HW + t] = (_Float16)y[v];
        }
    } else {
#pragma unroll
        for (int v = 0; v < 8; ++v) {
            const int m = m0 + v + 8 * h;
            const size_t idx = (size_t)m * CC + n0 + c;
            out32[idx] = y[v] + resid[idx];
        }
    }
}

// ---- fused QK^T + softmax ---------------------------------------------------
// 512 threads = 16 waves per block; block owns a full 16 x 4096 row strip.
// Wave w owns columns [w*256, w*256+256) as 16 WMMA accumulators (128 VGPRs).
// Softmax over the full row is done in registers + shfl + LDS cross-wave
// reduction; probabilities are written to HBM exactly once.
__global__ void __launch_bounds__(512)
qk_softmax_kernel(const _Float16* __restrict__ qn,
                  const _Float16* __restrict__ kn,
                  float* __restrict__ cor) {
    const int tid = threadIdx.x;
    const int wave = tid >> 5, lane = tid & 31;
    const int h = (lane >> 4) & 1, c = lane & 15;
    const int b  = blockIdx.y;
    const int m0 = blockIdx.x * 16;
    const _Float16* q = qn + (size_t)b * HW * CC;
    const _Float16* k = kn + (size_t)b * HW * CC;
    float* S = cor + (size_t)b * HW * HW;

    __shared__ _Float16 As[16 * CC];   // 8 KB: 16x256 A tile
    __shared__ float red[16][16];      // [wave][row] partials

    {   // cooperative A load: 4096 f16 / 512 threads = 8 each
        const int idx = tid * 8;
        const int r = idx >> 8, col = idx & (CC - 1);
        *(v8h*)(As + idx) = *(const v8h*)(q + (size_t)(m0 + r) * CC + col);
    }
    __syncthreads();

    v8f acc[16];
#pragma unroll
    for (int t = 0; t < 16; ++t) acc[t] = (v8f){};

#pragma unroll
    for (int k0 = 0; k0 < CC; k0 += 32) {
        v16h a = load_a_f16(As, CC, 0, k0, lane);
#pragma unroll
        for (int t = 0; t < 16; ++t) {
            v16h bb = load_bT_f16(k, CC, wave * 256 + t * 16, k0, lane);
            acc[t] = wmma16(a, bb, acc[t]);
        }
    }

    // ---- row max: over 16 tiles, then 16 lanes, then 16 waves
    float mx[8];
#pragma unroll
    for (int v = 0; v < 8; ++v) {
        float m = acc[0][v];
#pragma unroll
        for (int t = 1; t < 16; ++t) m = fmaxf(m, acc[t][v]);
        mx[v] = m;
    }
#pragma unroll
    for (int s = 1; s <= 8; s <<= 1)
#pragma unroll
        for (int v = 0; v < 8; ++v) mx[v] = fmaxf(mx[v], __shfl_xor(mx[v], s, 32));
    if (c == 0) {
#pragma unroll
        for (int v = 0; v < 8; ++v) red[wave][v + 8 * h] = mx[v];
    }
    __syncthreads();
#pragma unroll
    for (int v = 0; v < 8; ++v) {
        float m = red[0][v + 8 * h];
#pragma unroll
        for (int wv = 1; wv < 16; ++wv) m = fmaxf(m, red[wv][v + 8 * h]);
        mx[v] = m;
    }
    __syncthreads();   // before reusing red[] for the sums

    // ---- exponentiate in registers + row sum
    float sum[8] = {0.f, 0.f, 0.f, 0.f, 0.f, 0.f, 0.f, 0.f};
#pragma unroll
    for (int t = 0; t < 16; ++t)
#pragma unroll
        for (int v = 0; v < 8; ++v) {
            const float e = __expf(acc[t][v] - mx[v]);
            acc[t][v] = e;
            sum[v] += e;
        }
#pragma unroll
    for (int s = 1; s <= 8; s <<= 1)
#pragma unroll
        for (int v = 0; v < 8; ++v) sum[v] += __shfl_xor(sum[v], s, 32);
    if (c == 0) {
#pragma unroll
        for (int v = 0; v < 8; ++v) red[wave][v + 8 * h] = sum[v];
    }
    __syncthreads();
    float inv[8];
#pragma unroll
    for (int v = 0; v < 8; ++v) {
        float tot = red[0][v + 8 * h];
#pragma unroll
        for (int wv = 1; wv < 16; ++wv) tot += red[wv][v + 8 * h];
        inv[v] = 1.f / tot;
    }

    // ---- single write of probabilities
#pragma unroll
    for (int t = 0; t < 16; ++t) {
        const int n = wave * 256 + t * 16 + c;
#pragma unroll
        for (int v = 0; v < 8; ++v)
            S[(size_t)(m0 + v + 8 * h) * HW + n] = acc[t][v] * inv[v];
    }
}

// ---- O = P (f32->f16) @ vp ; vp supplied transposed [C x HW] f16 ----------
// 512 threads = 16 waves cover all 256 output channels -> P is read once.
// P chunks are converted to f16 once into a double-buffered LDS tile.
__global__ void __launch_bounds__(512)
av_kernel(const float* __restrict__ cor,
          const _Float16* __restrict__ vpT,
          float* __restrict__ O) {
    const int tid = threadIdx.x;
    const int wave = tid >> 5, lane = tid & 31;
    const int h = (lane >> 4) & 1, c = lane & 15;
    const int b  = blockIdx.y;
    const int m0 = blockIdx.x * 16;
    const int n0 = wave * 16;
    const float*    P = cor + (size_t)b * HW * HW;
    const _Float16* V = vpT + (size_t)b * CC * HW;

    __shared__ _Float16 As[2][16 * 128];   // double-buffered 16x128 f16, 8 KB

    auto stage = [&](int buf, int kb) {
        // 2048 elements / 512 threads = 4 each (one float4 load + 4 cvt)
        const int idx = tid * 4;
        const int r = idx >> 7, col = idx & 127;
        const float4 f = *(const float4*)(P + (size_t)(m0 + r) * HW + kb + col);
        _Float16* d = &As[buf][idx];
        d[0] = (_Float16)f.x; d[1] = (_Float16)f.y;
        d[2] = (_Float16)f.z; d[3] = (_Float16)f.w;
    };

    stage(0, 0);
    v8f acc = {};
    int buf = 0;
    for (int kb = 0; kb < HW; kb += 128) {
        __syncthreads();                       // stage(buf) visible to all
        if (kb + 128 < HW) {
            __builtin_prefetch(P + (size_t)(m0 + (tid & 15)) * HW + kb + 256, 0, 1);
            stage(buf ^ 1, kb + 128);          // fill other buffer async
        }
#pragma unroll
        for (int k0 = 0; k0 < 128; k0 += 32) {
            v16h a  = load_a_f16(As[buf], 128, 0, k0, lane);
            v16h bb = load_bT_f16(V, HW, n0, kb + k0, lane);
            acc = wmma16(a, bb, acc);
        }
        buf ^= 1;
    }
    float* Ob = O + (size_t)b * HW * CC;
#pragma unroll
    for (int v = 0; v < 8; ++v)
        Ob[(size_t)(m0 + v + 8 * h) * CC + n0 + c] = acc[v];
}

extern "C" void kernel_launch(void* const* d_in, const int* in_sizes, int n_in,
                              void* d_out, int out_size, void* d_ws, size_t ws_size,
                              hipStream_t stream) {
    (void)in_sizes; (void)n_in; (void)out_size; (void)ws_size;
    const float* query = (const float*)d_in[0];
    const float* ep    = (const float*)d_in[1];
    const float* value = (const float*)d_in[2];
    const float* wg = (const float*)d_in[3];
    const float* bg = (const float*)d_in[4];
    const float* wt = (const float*)d_in[5];
    const float* bt = (const float*)d_in[6];
    const float* wa = (const float*)d_in[7];
    const float* ba = (const float*)d_in[8];
    const float* wo = (const float*)d_in[9];
    const float* bo = (const float*)d_in[10];

    float* out0   = (float*)d_out;                          // query + out
    float* out_vp = out0 + (size_t)MTOT * CC;               // vp
    float* cor    = out_vp + (size_t)MTOT * CC;             // cor_map (probs)

    char* ws = (char*)d_ws;
    _Float16* qn16 = (_Float16*)(ws);                       // 16 MB [M][C] f16
    _Float16* kn16 = (_Float16*)(ws + (16ull << 20));       // 16 MB
    _Float16* vpT  = (_Float16*)(ws + (32ull << 20));       // 16 MB [B][C][HW] f16
    float*    o32  = (float*)   (ws + (48ull << 20));       // 32 MB [M][C] f32

    const dim3 pg(MTOT / 16), pb(512);
    proj_kernel<0><<<pg, pb, 0, stream>>>(query, wg, bg, qn16, nullptr, nullptr);
    proj_kernel<0><<<pg, pb, 0, stream>>>(ep,    wt, bt, kn16, nullptr, nullptr);
    proj_kernel<1><<<pg, pb, 0, stream>>>(value, wa, ba, vpT,  out_vp, nullptr);

    qk_softmax_kernel<<<dim3(HW / 16, NB), 512, 0, stream>>>(qn16, kn16, cor);
    av_kernel<<<dim3(HW / 16, NB), 512, 0, stream>>>(cor, vpT, o32);

    proj_kernel<2><<<pg, pb, 0, stream>>>(o32, wo, bo, nullptr, out0, query);
}
